// CausalSelfAttention_29111288332486
// MI455X (gfx1250) — matrix operand order repaired and checked by reading, not compiled, not run on hardware
//
#include <hip/hip_runtime.h>
#include <hip/hip_bf16.h>

// CausalSelfAttention forward for MI455X (gfx1250, wave32, WMMA).
// B=4, T=2048, C=1024, H=16, D=64.
// Pipeline: [bf16 QKV GEMM] -> [flash attention, WMMA+online softmax] -> [bf16 proj GEMM]
// Workspace (bf16): Q (B,H,T,D) | K (B,H,T,D) | V transposed (B,H,D,T) | y (B,T,C)

typedef __attribute__((ext_vector_type(16))) __bf16 v16bf;
typedef __attribute__((ext_vector_type(8)))  float  v8f;

union BFrag {
  v16bf v;
  unsigned short u[16];
  uint4 q[2];
};

// ---- f32 -> bf16 conversion ------------------------------------------------
// Prefer the clang builtin; otherwise use V_CVT_PK_BF16_F32 directly (CDNA5
// has packed-BF16 VALU support). One VALU op converts two floats.
#if __has_builtin(__builtin_amdgcn_cvt_pk_bf16_f32)
typedef __attribute__((ext_vector_type(2))) __bf16 v2bf;
__device__ inline unsigned int pack2bf(float lo, float hi) {
  union { v2bf v; unsigned int u; } c;
  c.v = __builtin_amdgcn_cvt_pk_bf16_f32(lo, hi);
  return c.u;
}
#else
__device__ inline unsigned int pack2bf(float lo, float hi) {
  unsigned int r;
  asm("v_cvt_pk_bf16_f32 %0, %1, %2" : "=v"(r) : "v"(lo), "v"(hi));
  return r;
}
#endif
__device__ inline unsigned short f2bf(float f) {
  return (unsigned short)pack2bf(f, 0.f);
}

__device__ inline float rmax16(float v) {
  #pragma unroll
  for (int o = 1; o < 16; o <<= 1) v = fmaxf(v, __shfl_xor(v, o, 32));
  return v;
}
__device__ inline float rsum16(float v) {
  #pragma unroll
  for (int o = 1; o < 16; o <<= 1) v += __shfl_xor(v, o, 32);
  return v;
}

// ---------------------------------------------------------------------------
// Shared GEMM mainloop: 128x128 block tile, K-step 32, 256 threads (8 waves).
// Wave w computes a 64x32 sub-tile: rows (w&1)*64, cols (w>>1)*32 -> 8 accums.
// A staged row-major [128][32] bf16; B staged transposed [128(n)][32(k)] bf16.
// ---------------------------------------------------------------------------
template <bool A_IS_BF16>
__device__ inline void gemm_mainloop(const void* gA_, const float* gB,
                                     int K, int N, int blockRow, int blockCol,
                                     unsigned short* ldsA, unsigned short* ldsBT,
                                     v8f acc[8]) {
  const int tid  = threadIdx.x;
  const int lane = tid & 31, wave = tid >> 5;
  const int half = lane >> 4, l16 = lane & 15;
  const int waveM = (wave & 1) * 64;
  const int waveN = (wave >> 1) * 32;

  v8f zero = {};
  #pragma unroll
  for (int i = 0; i < 8; ++i) acc[i] = zero;

  const int nkt = K >> 5;
  for (int kt = 0; kt < nkt; ++kt) {
    const int k0 = kt << 5;
    __syncthreads();

    // ---- Stage A tile (128 rows x 32 cols) as bf16, row-major ----
    if (A_IS_BF16) {
      const unsigned short* gA = (const unsigned short*)gA_;
      #pragma unroll
      for (int j = 0; j < 2; ++j) {
        const int i8 = tid + j * 256;            // 0..511 chunks of 8
        const int m = i8 >> 2, k8 = (i8 & 3) * 8;
        *(uint4*)(ldsA + m * 32 + k8) =
            *(const uint4*)(gA + (size_t)(blockRow + m) * K + k0 + k8);
      }
    } else {
      const float* gA = (const float*)gA_;
      #pragma unroll
      for (int j = 0; j < 4; ++j) {
        const int i4 = tid + j * 256;            // 0..1023 chunks of 4
        const int m = i4 >> 3, k4 = (i4 & 7) * 4;
        const float4 f = *(const float4*)(gA + (size_t)(blockRow + m) * K + k0 + k4);
        uint2 p; p.x = pack2bf(f.x, f.y); p.y = pack2bf(f.z, f.w);
        *(uint2*)(ldsA + m * 32 + k4) = p;
      }
    }

    // ---- Stage B tile (32 x 128) transposed -> ldsBT[n][k] ----
    #pragma unroll
    for (int j = 0; j < 4; ++j) {
      const int i4 = tid + j * 256;              // 0..1023 chunks of 4
      const int kk = i4 >> 5, n4 = (i4 & 31) * 4;
      const float4 f = *(const float4*)(gB + (size_t)(k0 + kk) * N + blockCol + n4);
      const unsigned int u01 = pack2bf(f.x, f.y);
      const unsigned int u23 = pack2bf(f.z, f.w);
      ldsBT[(n4 + 0) * 32 + kk] = (unsigned short)u01;
      ldsBT[(n4 + 1) * 32 + kk] = (unsigned short)(u01 >> 16);
      ldsBT[(n4 + 2) * 32 + kk] = (unsigned short)u23;
      ldsBT[(n4 + 3) * 32 + kk] = (unsigned short)(u23 >> 16);
    }

    // Prefetch next k-tile into near caches (speculative; OOB dropped).
    __builtin_prefetch((const char*)gB + ((size_t)(k0 + 32 + (tid >> 3)) * N + blockCol + (tid & 7) * 16) * 4, 0, 3);
    if (!A_IS_BF16)
      __builtin_prefetch((const char*)gA_ + ((size_t)(blockRow + (tid >> 1)) * K + k0 + 32 + (tid & 1) * 16) * 4, 0, 3);

    __syncthreads();

    BFrag a[4], b[2];
    #pragma unroll
    for (int mt = 0; mt < 4; ++mt) {
      const unsigned short* p = ldsA + (waveM + mt * 16 + l16) * 32;
      a[mt].q[0] = *(const uint4*)(p + 16 * half);       // e=0..7  : k = 16*half + e
      a[mt].q[1] = *(const uint4*)(p + 16 * half + 8);   // e=8..15 : k = 16*half + e
    }
    #pragma unroll
    for (int nt = 0; nt < 2; ++nt) {
      const unsigned short* p = ldsBT + (waveN + nt * 16 + l16) * 32 + 16 * half;
      b[nt].q[0] = *(const uint4*)(p);                   // e=0..15 : k = 16*half + e
      b[nt].q[1] = *(const uint4*)(p + 8);
    }
    #pragma unroll
    for (int mt = 0; mt < 4; ++mt)
      #pragma unroll
      for (int nt = 0; nt < 2; ++nt)
        acc[mt * 2 + nt] = __builtin_amdgcn_wmma_f32_16x16x32_bf16(
            false, a[mt].v, false, b[nt].v, (short)0, acc[mt * 2 + nt], false, false);
  }
}

// ---------------------------------------------------------------------------
// Kernel 1: QKV projection.  x (8192x1024) @ w_qkv (1024x3072) -> Q,K,V bf16.
// ---------------------------------------------------------------------------
__global__ __launch_bounds__(256) void qkv_gemm_kernel(
    const float* __restrict__ x, const float* __restrict__ w,
    unsigned short* __restrict__ qws, unsigned short* __restrict__ kws,
    unsigned short* __restrict__ vws) {
  __shared__ __align__(16) unsigned short ldsA[128 * 32];
  __shared__ __align__(16) unsigned short ldsBT[128 * 32];
  v8f acc[8];
  const int blockRow = blockIdx.y * 128, blockCol = blockIdx.x * 128;
  gemm_mainloop<false>(x, w, 1024, 3072, blockRow, blockCol, ldsA, ldsBT, acc);

  const int lane = threadIdx.x & 31, wave = threadIdx.x >> 5;
  const int half = lane >> 4, l16 = lane & 15;
  const int waveM = (wave & 1) * 64, waveN = (wave >> 1) * 32;

  // A 16-aligned 16-wide column window never straddles a 64- or 1024-boundary,
  // so `which`, `hh`, base `dd` are wave-uniform per nt tile.
  #pragma unroll
  for (int nt = 0; nt < 2; ++nt) {
    const int colBase = blockCol + waveN + nt * 16;
    const int which = colBase >> 10;          // 0=Q, 1=K, 2=V (uniform)
    const int cc0 = colBase & 1023;
    const int hh = cc0 >> 6;                  // head (uniform)
    const int dd = (cc0 & 63) + l16;          // 0..63
    if (which < 2) {
      unsigned short* dstQK = (which == 0) ? qws : kws;
      #pragma unroll
      for (int mt = 0; mt < 4; ++mt)
        #pragma unroll
        for (int r = 0; r < 8; ++r) {
          const int gr = blockRow + waveM + mt * 16 + r + 8 * half;   // token
          const int bb = gr >> 11, t = gr & 2047;
          dstQK[((size_t)(bb * 16 + hh) * 2048 + t) * 64 + dd] =
              f2bf(acc[mt * 2 + nt][r]);
        }
    } else {
      // Transposed V: the 8 accumulator rows map to 8 consecutive t addresses
      // (8-aligned, never crossing the T=2048 row) -> one uint4 store per tile.
      #pragma unroll
      for (int mt = 0; mt < 4; ++mt) {
        const int gr0 = blockRow + waveM + mt * 16 + 8 * half;
        const int bb = gr0 >> 11, t0 = gr0 & 2047;
        const v8f a = acc[mt * 2 + nt];
        uint4 pk;
        pk.x = pack2bf(a[0], a[1]); pk.y = pack2bf(a[2], a[3]);
        pk.z = pack2bf(a[4], a[5]); pk.w = pack2bf(a[6], a[7]);
        *(uint4*)(vws + ((size_t)(bb * 16 + hh) * 64 + dd) * 2048 + t0) = pk;
      }
    }
  }
}

// ---------------------------------------------------------------------------
// Kernel 2: flash attention. One wave per 16-query tile; 32-key tiles.
// ---------------------------------------------------------------------------
__global__ __launch_bounds__(128) void attn_kernel(
    const unsigned short* __restrict__ qws, const unsigned short* __restrict__ kws,
    const unsigned short* __restrict__ vws, unsigned short* __restrict__ yws) {
  __shared__ __align__(16) unsigned short ldsP[4][16 * 32];   // per-wave P staging

  const int lane = threadIdx.x & 31, wave = threadIdx.x >> 5;
  const int half = lane >> 4, l16 = lane & 15;
  const int qt = blockIdx.x * 4 + wave;
  const int h = blockIdx.y, b = blockIdx.z;
  const int qbase = qt * 16;
  const size_t bh = (size_t)(b * 16 + h);
  const unsigned short* Qp    = qws + (bh * 2048 + qbase) * 64;
  const unsigned short* Kbase = kws + bh * 2048 * 64;
  const unsigned short* Vbase = vws + bh * 64 * 2048;
  unsigned short* lp = ldsP[wave];

  // Q tile (16x64) as two A-fragments (d-chunks of 32); lives in regs all loop.
  BFrag qa[2];
  #pragma unroll
  for (int ks = 0; ks < 2; ++ks) {
    const unsigned short* p = Qp + l16 * 64 + ks * 32;
    qa[ks].q[0] = *(const uint4*)(p + 16 * half);
    qa[ks].q[1] = *(const uint4*)(p + 16 * half + 8);
  }

  float M[8], L[8];
  v8f O[4];
  v8f zero = {};
  #pragma unroll
  for (int r = 0; r < 8; ++r) { M[r] = -1e30f; L[r] = 0.f; }
  #pragma unroll
  for (int c = 0; c < 4; ++c) O[c] = zero;

  const float scale = 0.125f;                   // 1/sqrt(64)
  const int ktend = (qbase + 15) >> 5;
  for (int kt = 0; kt <= ktend; ++kt) {
    const int kb = kt * 32;
    const unsigned short* Kp = Kbase + (size_t)kb * 64;

    // S (16x32) = Q @ K^T : two 16x16 C-tiles, accumulated over two d-chunks.
    v8f S0 = zero, S1 = zero;
    #pragma unroll
    for (int ks = 0; ks < 2; ++ks) {
      BFrag b0, b1;
      const unsigned short* p0 = Kp + l16 * 64 + ks * 32 + 16 * half;
      b0.q[0] = *(const uint4*)p0;       b0.q[1] = *(const uint4*)(p0 + 8);
      const unsigned short* p1 = Kp + (16 + l16) * 64 + ks * 32 + 16 * half;
      b1.q[0] = *(const uint4*)p1;       b1.q[1] = *(const uint4*)(p1 + 8);
      S0 = __builtin_amdgcn_wmma_f32_16x16x32_bf16(false, qa[ks].v, false, b0.v, (short)0, S0, false, false);
      S1 = __builtin_amdgcn_wmma_f32_16x16x32_bf16(false, qa[ks].v, false, b1.v, (short)0, S1, false, false);
    }

    // Scale + causal mask + online softmax (row m = r + 8*half, col n = l16).
    float p0v[8], p1v[8], alpha[8];
    #pragma unroll
    for (int r = 0; r < 8; ++r) {
      const int qg = qbase + r + 8 * half;
      float s0 = S0[r] * scale; if (kb + l16 > qg)      s0 = -1e30f;
      float s1 = S1[r] * scale; if (kb + 16 + l16 > qg) s1 = -1e30f;
      const float mt = rmax16(fmaxf(s0, s1));
      const float Mn = fmaxf(M[r], mt);
      alpha[r] = __expf(M[r] - Mn);
      M[r] = Mn;
      p0v[r] = __expf(s0 - Mn);
      p1v[r] = __expf(s1 - Mn);
      const float rs = rsum16(p0v[r] + p1v[r]);
      L[r] = L[r] * alpha[r] + rs;
    }
    #pragma unroll
    for (int c = 0; c < 4; ++c)
      #pragma unroll
      for (int r = 0; r < 8; ++r) O[c][r] *= alpha[r];

    // Stage P (C-layout) -> LDS -> reload as A-fragment (16x32 bf16).
    // LDS ops within one wave are in-order; compiler inserts s_wait_dscnt.
    #pragma unroll
    for (int r = 0; r < 8; ++r) {
      const unsigned int pk = pack2bf(p0v[r], p1v[r]);
      lp[(r + 8 * half) * 32 + l16]      = (unsigned short)pk;
      lp[(r + 8 * half) * 32 + 16 + l16] = (unsigned short)(pk >> 16);
    }
    BFrag pa;
    pa.q[0] = *(const uint4*)(lp + l16 * 32 + 16 * half);
    pa.q[1] = *(const uint4*)(lp + l16 * 32 + 16 * half + 8);

    // O (16x64) += P (16x32) @ V (32x64); V is stored transposed (B,H,D,T)
    // so B-fragments are contiguous 32-byte loads along T.
    #pragma unroll
    for (int c = 0; c < 4; ++c) {
      BFrag vb;
      const unsigned short* vp = Vbase + (size_t)(c * 16 + l16) * 2048 + kb + 16 * half;
      vb.q[0] = *(const uint4*)vp;   vb.q[1] = *(const uint4*)(vp + 8);
      O[c] = __builtin_amdgcn_wmma_f32_16x16x32_bf16(false, pa.v, false, vb.v, (short)0, O[c], false, false);
    }
  }

  // Normalize and write y (B,T,C) bf16.
  #pragma unroll
  for (int r = 0; r < 8; ++r) {
    const float inv = 1.0f / L[r];
    const int t = qbase + r + 8 * half;
    #pragma unroll
    for (int c = 0; c < 4; ++c)
      yws[((size_t)b * 2048 + t) * 1024 + h * 64 + c * 16 + l16] = f2bf(O[c][r] * inv);
  }
}

// ---------------------------------------------------------------------------
// Kernel 3: output projection. y (8192x1024 bf16) @ w_proj (1024x1024) -> f32.
// ---------------------------------------------------------------------------
__global__ __launch_bounds__(256) void proj_gemm_kernel(
    const unsigned short* __restrict__ y, const float* __restrict__ w,
    float* __restrict__ out) {
  __shared__ __align__(16) unsigned short ldsA[128 * 32];
  __shared__ __align__(16) unsigned short ldsBT[128 * 32];
  v8f acc[8];
  const int blockRow = blockIdx.y * 128, blockCol = blockIdx.x * 128;
  gemm_mainloop<true>(y, w, 1024, 1024, blockRow, blockCol, ldsA, ldsBT, acc);

  const int lane = threadIdx.x & 31, wave = threadIdx.x >> 5;
  const int half = lane >> 4, l16 = lane & 15;
  const int waveM = (wave & 1) * 64, waveN = (wave >> 1) * 32;
  #pragma unroll
  for (int mt = 0; mt < 4; ++mt)
    #pragma unroll
    for (int nt = 0; nt < 2; ++nt)
      #pragma unroll
      for (int r = 0; r < 8; ++r) {
        const int gr = blockRow + waveM + mt * 16 + r + 8 * half;
        const int gc = blockCol + waveN + nt * 16 + l16;
        out[(size_t)gr * 1024 + gc] = acc[mt * 2 + nt][r];
      }
}

// ---------------------------------------------------------------------------
extern "C" void kernel_launch(void* const* d_in, const int* in_sizes, int n_in,
                              void* d_out, int out_size, void* d_ws, size_t ws_size,
                              hipStream_t stream) {
  const float* x      = (const float*)d_in[0];  // (4,2048,1024)
  const float* w_qkv  = (const float*)d_in[1];  // (1024,3072)
  const float* w_proj = (const float*)d_in[2];  // (1024,1024)

  const size_t E = (size_t)4 * 16 * 2048 * 64;  // 8,388,608 bf16 elements
  unsigned short* qws = (unsigned short*)d_ws;
  unsigned short* kws = qws + E;
  unsigned short* vws = kws + E;
  unsigned short* yws = vws + E;

  // 1) QKV projection: (8192x1024) @ (1024x3072)
  qkv_gemm_kernel<<<dim3(3072 / 128, 8192 / 128), 256, 0, stream>>>(x, w_qkv, qws, kws, vws);

  // 2) Flash attention: 128 q-tiles x 16 heads x 4 batch; 4 waves per block
  attn_kernel<<<dim3(2048 / 64, 16, 4), 128, 0, stream>>>(qws, kws, vws, yws);

  // 3) Output projection: (8192x1024) @ (1024x1024) -> fp32 out
  proj_gemm_kernel<<<dim3(1024 / 128, 8192 / 128), 256, 0, stream>>>(yws, w_proj, (float*)d_out);
}
